// MultilayerGRU_13125420057296
// MI455X (gfx1250) — matrix operand order repaired and checked by reading, not compiled, not run on hardware
//
#include <hip/hip_runtime.h>

// ---------------------------------------------------------------------------
// Multilayer GRU for MI455X (gfx1250), wave32 + WMMA bf16.
//   B=32, S=512, H=1024, L=2, O=1024
//   prep  : weights -> bf16 packed B layout; x -> bf16 rows; h0 -> f32+bf16
//   xproj : x_bf16 @ Wx0^T + bx0 -> xp0[s*32+b][3H] bf16   (parallel)
//   steps : persistent kernel, 512 steps x 4 phases, grid barrier between;
//           dual-M-tile waves: each B fragment loaded once, 2 WMMAs issued
//   out   : h1hist(bf16) @ Wo^T + bo -> (B,S,O)            (parallel)
// ---------------------------------------------------------------------------

typedef __attribute__((ext_vector_type(16))) __bf16 v16bf;
typedef __attribute__((ext_vector_type(8)))  float  v8f;

constexpr int    kB  = 32;
constexpr int    kS  = 512;
constexpr int    kH  = 1024;
constexpr int    kO  = 1024;
constexpr size_t kHH = (size_t)kH * kH;          // 1 Mi elements
constexpr int    kBH = kB * kH;                  // 32768

// packed-weight matrix ids (each kHH bf16 elements):
//   0..2 : Wh[0][g]   3..5 : Wx[1][g]   6..8 : Wh[1][g]   9 : Wo   10..12 : Wx[0][g]
constexpr size_t TILE_SHORTS = 512;              // one 16(N)x32(K) bf16 tile
constexpr size_t COL_SHORTS  = 32 * TILE_SHORTS; // 32 k-tiles (K=1024) per col tile

// workspace layout (bytes)
constexpr size_t OFF_BAR = 0;
constexpr size_t OFF_PW  = 256;                                  // 13 matrices bf16
constexpr size_t SZ_PW   = 13 * kHH * 2;
constexpr size_t OFF_XBF = OFF_PW + SZ_PW;                       // [S*B][H] bf16
constexpr size_t SZ_XBF  = (size_t)kS * kB * kH * 2;
constexpr size_t OFF_XP0 = OFF_XBF + SZ_XBF;                     // [S*B][3H] bf16
constexpr size_t SZ_XP0  = (size_t)kS * kB * 3 * kH * 2;
constexpr size_t OFF_H1H = OFF_XP0 + SZ_XP0;                     // [S*B][H] bf16
constexpr size_t SZ_H1H  = (size_t)kS * kB * kH * 2;
constexpr size_t OFF_HF  = OFF_H1H + SZ_H1H;                     // [2][B][H] f32
constexpr size_t OFF_HB  = OFF_HF + 2 * (size_t)kBH * 4;         // [2][B][H] bf16
constexpr size_t OFF_RHB = OFF_HB + 2 * (size_t)kBH * 2;         // [2][B][H] bf16
constexpr size_t OFF_Z   = OFF_RHB + 2 * (size_t)kBH * 2;        // [2][B][H] f32
constexpr size_t OFF_G2  = OFF_Z + 2 * (size_t)kBH * 4;          // [B][H]    f32

// ---------------- helpers ----------------

__device__ __forceinline__ unsigned short f2bf(float x) {
  union { float f; unsigned u; } v; v.f = x;
  unsigned r = v.u + 0x7FFFu + ((v.u >> 16) & 1u);   // round-to-nearest-even
  return (unsigned short)(r >> 16);
}
__device__ __forceinline__ float bf2f(unsigned short s) {
  union { float f; unsigned u; } v; v.u = ((unsigned)s) << 16; return v.f;
}

// A-operand 16-bit layout (ISA 7.12.2): lane L holds row M=L&15, half=L>>4.
// Vector elems 0..7 cover K = kbase + 8*half + {0..7}; elems 8..15 cover
// K = kbase + 16 + 8*half + {0..7}  -> two contiguous b128 loads per lane.
__device__ __forceinline__ v16bf load_a_bf16(const unsigned short* p, int half) {
  union { v16bf v; uint4 q[2]; } f;
  f.q[0] = *(const uint4*)(p + half * 8);
  f.q[1] = *(const uint4*)(p + 16 + half * 8);
  return f.v;
}

// B-operand: pre-packed so lane L reads 16 contiguous bf16 (two b128 loads).
__device__ __forceinline__ v16bf load_b(const unsigned short* tile, int lane) {
  union { v16bf v; uint4 q[2]; } f;
  const uint4* p = (const uint4*)(tile + lane * 16);
  f.q[0] = p[0];
  f.q[1] = p[1];
  return f.v;
}

__device__ __forceinline__ v8f wmma_bf16(v16bf a, v16bf b, v8f c) {
  return __builtin_amdgcn_wmma_f32_16x16x32_bf16(false, a, false, b, (short)0, c,
                                                 false, false);
}

// dual-M 32x16 output block: rows [mbase, mbase+32), one B load -> two WMMAs
__device__ __forceinline__ void tile_gemm2(const unsigned short* A, int lda, int mbase,
                                           const unsigned short* Bp, int nkt,
                                           int lane, int half, v8f& c0, v8f& c1) {
  const unsigned short* row0 = A + (size_t)(mbase + (lane & 15)) * lda;
  const unsigned short* row1 = row0 + (size_t)16 * lda;
#pragma unroll 8
  for (int kt = 0; kt < nkt; ++kt) {
    v16bf b  = load_b(Bp + kt * TILE_SHORTS, lane);
    v16bf a0 = load_a_bf16(row0 + kt * 32, half);
    v16bf a1 = load_a_bf16(row1 + kt * 32, half);
    c0 = wmma_bf16(a0, b, c0);
    c1 = wmma_bf16(a1, b, c1);
  }
}

__device__ __forceinline__ float sigmoidf_fast(float x) {
  return 1.0f / (1.0f + __expf(-x));
}

// monotonic-counter grid barrier (only 64 blocks launched -> co-resident)
__device__ __forceinline__ void grid_sync(unsigned* bar, unsigned epoch, unsigned nb) {
  __threadfence();
  __syncthreads();
  if (threadIdx.x == 0) {
    __hip_atomic_fetch_add(bar, 1u, __ATOMIC_RELEASE, __HIP_MEMORY_SCOPE_AGENT);
    const unsigned target = epoch * nb;
    while (__hip_atomic_load(bar, __ATOMIC_ACQUIRE, __HIP_MEMORY_SCOPE_AGENT) < target)
      __builtin_amdgcn_s_sleep(2);
  }
  __syncthreads();
  __threadfence();
}

// ---------------- kernels ----------------

// pack weights -> bf16 B layout; x -> bf16 rows [s*32+b][H]; h0 -> f32+bf16; bar=0
// B tile element (lane, j) = W[nt*16 + (lane&15)][kt*32 + (j&7) + 8*(lane>>4) + 16*(j>>3)]
__global__ void __launch_bounds__(256) gru_prep(const float* __restrict__ Wx,
                                                const float* __restrict__ Wh,
                                                const float* __restrict__ Wo,
                                                const float* __restrict__ x,
                                                const float* __restrict__ h0,
                                                unsigned short* __restrict__ pw,
                                                unsigned short* __restrict__ xbf,
                                                float* __restrict__ hf32,
                                                unsigned short* __restrict__ hbf,
                                                unsigned* __restrict__ bar) {
  const size_t i0     = (size_t)blockIdx.x * blockDim.x + threadIdx.x;
  const size_t stride = (size_t)gridDim.x * blockDim.x;
  if (i0 == 0) *bar = 0u;
  const size_t NW = 13 * kHH;                    // weight elements
  const size_t NX = (size_t)kS * kB * kH;        // x elements
  for (size_t t = i0; t < NW + NX + (size_t)2 * kBH; t += stride) {
    if (t < NW) {
      const int    m = (int)(t / kHH);
      const size_t e = t % kHH;
      const int j    = (int)(e & 15);
      const int lane = (int)((e >> 4) & 31);
      const int kt   = (int)((e >> 9) & 31);
      const int nt   = (int)(e >> 14);
      const int n    = nt * 16 + (lane & 15);
      const int k    = kt * 32 + (j & 7) + ((lane >> 4) << 3) + ((j >> 3) << 4);
      const float* src;
      if      (m < 3)  src = Wh + (size_t)m * kHH;           // Wh[0][g]
      else if (m < 6)  src = Wx + (size_t)(3 + m - 3) * kHH; // Wx[1][g]
      else if (m < 9)  src = Wh + (size_t)(3 + m - 6) * kHH; // Wh[1][g]
      else if (m == 9) src = Wo;                             // Wo
      else             src = Wx + (size_t)(m - 10) * kHH;    // Wx[0][g]
      pw[t] = f2bf(src[(size_t)n * kH + k]);
    } else if (t < NW + NX) {
      const size_t e = t - NW;                   // R = s*32+b rows, k fastest
      const int R = (int)(e >> 10), k = (int)(e & 1023);
      const int s = R >> 5, b = R & 31;
      xbf[e] = f2bf(x[(size_t)b * kS * kH + (size_t)s * kH + k]);
    } else {
      const size_t e = t - NW - NX;              // h0: (b, l, k) -> [l][b][k]
      const int b = (int)(e >> 11), l = (int)((e >> 10) & 1), k = (int)(e & 1023);
      const float v = h0[e];
      hf32[(size_t)l * kBH + b * kH + k] = v;
      hbf [(size_t)l * kBH + b * kH + k] = f2bf(v);
    }
  }
}

// xp0[s*32+b][g*H+n] = bf16( x[b,s,:] . Wx0[g][n,:] + bx0[g][n] )
__global__ void __launch_bounds__(256) gru_xproj(const unsigned short* __restrict__ xbf,
                                                 const unsigned short* __restrict__ pw,
                                                 const float* __restrict__ bx,
                                                 unsigned short* __restrict__ xp0) {
  const int lane = threadIdx.x & 31, half = lane >> 4;
  const int wave   = blockIdx.x * (blockDim.x >> 5) + (threadIdx.x >> 5);
  const int nwaves = gridDim.x * (blockDim.x >> 5);
  for (int t = wave; t < 512 * 192; t += nwaves) {
    const int mt2 = t & 511, nt = t >> 9;        // 512 M-pairs x 192 N-tiles
    const int g = nt >> 6, ntg = nt & 63;
    v8f c0 = {}, c1 = {};
    tile_gemm2(xbf, kH, mt2 * 32,
               pw + (size_t)(10 + g) * kHH + (size_t)ntg * COL_SHORTS,
               32, lane, half, c0, c1);
    const int n = ntg * 16 + (lane & 15);
    const float bias = bx[g * kH + n];
#pragma unroll
    for (int r = 0; r < 8; ++r) {
      const int R0 = mt2 * 32 + r + half * 8;    // R = s*32 + b
      xp0[(size_t)R0 * (3 * kH) + g * kH + n]        = f2bf(c0[r] + bias);
      xp0[(size_t)(R0 + 16) * (3 * kH) + g * kH + n] = f2bf(c1[r] + bias);
    }
  }
}

// persistent sequential scan: 512 steps x 4 phases (M=32 covered by one wave)
__global__ void __launch_bounds__(256) gru_steps(const unsigned short* __restrict__ pw,
                                                 const unsigned short* __restrict__ xp0,
                                                 const float* __restrict__ bx,
                                                 float* __restrict__ hf32,          // [2][B][H] f32
                                                 unsigned short* __restrict__ hbf,  // [2][B][H] bf16
                                                 unsigned short* __restrict__ rhb,  // [2][B][H] bf16
                                                 float* __restrict__ zb,            // [2][B][H] f32
                                                 float* __restrict__ g2,            // [B][H]    f32
                                                 unsigned short* __restrict__ h1h,
                                                 float* __restrict__ out_hidden,
                                                 unsigned* __restrict__ bar) {
  const int lane = threadIdx.x & 31, half = lane >> 4;
  const int wave   = blockIdx.x * (blockDim.x >> 5) + (threadIdx.x >> 5);
  const int nwaves = gridDim.x * (blockDim.x >> 5);
  const unsigned nb = gridDim.x;
  unsigned epoch = 0;
  const int mrow0 = half * 8;                    // row of c[0] within 16-row tile

#pragma unroll 1
  for (int s = 0; s < kS; ++s) {
    const unsigned short* xrow = xp0 + (size_t)s * kB * 3 * kH;

    // ---- Phase 1: layer-0 z/r gates: M=32, N=2048, K=1024 (128 tasks)
    for (int t = wave; t < 128; t += nwaves) {
      const int g = t >> 6, ntg = t & 63;
      v8f c0 = {}, c1 = {};
      tile_gemm2(hbf, kH, 0, pw + (size_t)g * kHH + (size_t)ntg * COL_SHORTS,
                 32, lane, half, c0, c1);
      const int n = ntg * 16 + (lane & 15);
#pragma unroll
      for (int r = 0; r < 8; ++r) {
#pragma unroll
        for (int mhalf = 0; mhalf < 2; ++mhalf) {
          const int b = mhalf * 16 + r + mrow0;
          const float acc = mhalf ? c1[r] : c0[r];
          const float pre = acc + bf2f(xrow[(size_t)b * 3072 + g * kH + n]);
          const float a = sigmoidf_fast(pre);
          if (g == 0) zb[b * kH + n] = a;                           // z
          else        rhb[b * kH + n] = f2bf(a * hf32[b * kH + n]); // r * h_prev
        }
      }
    }
    grid_sync(bar, ++epoch, nb);

    // ---- Phase 2: layer-0 candidate + h update: N=1024 (64 tasks)
    for (int t = wave; t < 64; t += nwaves) {
      v8f c0 = {}, c1 = {};
      tile_gemm2(rhb, kH, 0, pw + (size_t)2 * kHH + (size_t)t * COL_SHORTS,
                 32, lane, half, c0, c1);
      const int n = t * 16 + (lane & 15);
#pragma unroll
      for (int r = 0; r < 8; ++r) {
#pragma unroll
        for (int mhalf = 0; mhalf < 2; ++mhalf) {
          const int b = mhalf * 16 + r + mrow0;
          const float acc = mhalf ? c1[r] : c0[r];
          const float cc = tanhf(acc + bf2f(xrow[(size_t)b * 3072 + 2 * kH + n]));
          const float z  = zb[b * kH + n];
          const float hp = hf32[b * kH + n];
          const float hn = z * hp + (1.0f - z) * cc;
          hf32[b * kH + n] = hn;
          hbf [b * kH + n] = f2bf(hn);
        }
      }
    }
    grid_sync(bar, ++epoch, nb);

    // ---- Phase 3: layer-1 input proj + z/r gates (192 tasks; K=2048 for g<2)
    for (int t = wave; t < 192; t += nwaves) {
      const int g = t >> 6, ntg = t & 63;
      v8f c0 = {}, c1 = {};
      tile_gemm2(hbf, kH, 0,                       // h_l0_new @ Wx1[g]^T
                 pw + (size_t)(3 + g) * kHH + (size_t)ntg * COL_SHORTS,
                 32, lane, half, c0, c1);
      if (g < 2)
        tile_gemm2(hbf + kBH, kH, 0,               // + h1_prev @ Wh1[g]^T
                   pw + (size_t)(6 + g) * kHH + (size_t)ntg * COL_SHORTS,
                   32, lane, half, c0, c1);
      const int n = ntg * 16 + (lane & 15);
      const float bias = bx[(3 + g) * kH + n];
#pragma unroll
      for (int r = 0; r < 8; ++r) {
#pragma unroll
        for (int mhalf = 0; mhalf < 2; ++mhalf) {
          const int b = mhalf * 16 + r + mrow0;
          const float pre = (mhalf ? c1[r] : c0[r]) + bias;
          if (g == 0)      zb[kBH + b * kH + n] = sigmoidf_fast(pre);
          else if (g == 1) rhb[kBH + b * kH + n] =
                             f2bf(sigmoidf_fast(pre) * hf32[kBH + b * kH + n]);
          else             g2[b * kH + n] = pre;   // candidate partial
        }
      }
    }
    grid_sync(bar, ++epoch, nb);

    // ---- Phase 4: layer-1 candidate recurrent + h update + history (64 tasks)
    for (int t = wave; t < 64; t += nwaves) {
      v8f c0 = {}, c1 = {};
      tile_gemm2(rhb + kBH, kH, 0, pw + (size_t)8 * kHH + (size_t)t * COL_SHORTS,
                 32, lane, half, c0, c1);
      const int n = t * 16 + (lane & 15);
#pragma unroll
      for (int r = 0; r < 8; ++r) {
#pragma unroll
        for (int mhalf = 0; mhalf < 2; ++mhalf) {
          const int b = mhalf * 16 + r + mrow0;
          const float cc = tanhf((mhalf ? c1[r] : c0[r]) + g2[b * kH + n]);
          const float z  = zb[kBH + b * kH + n];
          const float hp = hf32[kBH + b * kH + n];
          const float hn = z * hp + (1.0f - z) * cc;
          hf32[kBH + b * kH + n] = hn;
          const unsigned short hb16 = f2bf(hn);
          hbf[kBH + b * kH + n] = hb16;
          h1h[(size_t)(s * kB + b) * kH + n] = hb16;
        }
      }
    }
    grid_sync(bar, ++epoch, nb);
  }

  // final hidden state -> (B, L, H)
  const int tid = blockIdx.x * blockDim.x + threadIdx.x;
  for (int i = tid; i < 2 * kBH; i += gridDim.x * blockDim.x) {
    const int l = i >> 15, b = (i >> 10) & 31, k = i & 1023;
    out_hidden[(size_t)b * 2 * kH + l * kH + k] = hf32[i];
  }
}

// out[b,s,n] = h1hist[s*32+b,:] . Wo[n,:] + bo[n]
__global__ void __launch_bounds__(256) gru_out(const unsigned short* __restrict__ h1h,
                                               const unsigned short* __restrict__ pw,
                                               const float* __restrict__ bo,
                                               float* __restrict__ out) {
  const int lane = threadIdx.x & 31, half = lane >> 4;
  const int wave   = blockIdx.x * (blockDim.x >> 5) + (threadIdx.x >> 5);
  const int nwaves = gridDim.x * (blockDim.x >> 5);
  for (int t = wave; t < 512 * 64; t += nwaves) {
    const int mt2 = t & 511, nt = t >> 9;
    v8f c0 = {}, c1 = {};
    tile_gemm2(h1h, kH, mt2 * 32,
               pw + (size_t)9 * kHH + (size_t)nt * COL_SHORTS, 32, lane, half, c0, c1);
    const int n = nt * 16 + (lane & 15);
    const float bias = bo[n];
#pragma unroll
    for (int r = 0; r < 8; ++r) {
      const int R0 = mt2 * 32 + r + half * 8;
      const int s0 = R0 >> 5, b0 = R0 & 31;
      const int R1 = R0 + 16;
      const int s1 = R1 >> 5, b1 = R1 & 31;
      out[(size_t)b0 * kS * kO + (size_t)s0 * kO + n] = c0[r] + bias;
      out[(size_t)b1 * kS * kO + (size_t)s1 * kO + n] = c1[r] + bias;
    }
  }
}

// ---------------- launch ----------------

extern "C" void kernel_launch(void* const* d_in, const int* in_sizes, int n_in,
                              void* d_out, int out_size, void* d_ws, size_t ws_size,
                              hipStream_t stream) {
  (void)in_sizes; (void)n_in; (void)out_size; (void)ws_size;
  const float* x  = (const float*)d_in[0];
  const float* h0 = (const float*)d_in[1];
  const float* Wx = (const float*)d_in[2];
  const float* Wh = (const float*)d_in[3];
  const float* bx = (const float*)d_in[4];
  const float* Wo = (const float*)d_in[5];
  const float* bo = (const float*)d_in[6];

  char* ws = (char*)d_ws;
  unsigned*       bar  = (unsigned*)(ws + OFF_BAR);
  unsigned short* pw   = (unsigned short*)(ws + OFF_PW);
  unsigned short* xbf  = (unsigned short*)(ws + OFF_XBF);
  unsigned short* xp0  = (unsigned short*)(ws + OFF_XP0);
  unsigned short* h1h  = (unsigned short*)(ws + OFF_H1H);
  float*          hf32 = (float*)(ws + OFF_HF);
  unsigned short* hbf  = (unsigned short*)(ws + OFF_HB);
  unsigned short* rhb  = (unsigned short*)(ws + OFF_RHB);
  float*          zb   = (float*)(ws + OFF_Z);
  float*          g2   = (float*)(ws + OFF_G2);

  float* out        = (float*)d_out;
  float* out_hidden = out + (size_t)kB * kS * kO;

  gru_prep <<<4096, 256, 0, stream>>>(Wx, Wh, Wo, x, h0, pw, xbf, hf32, hbf, bar);
  gru_xproj<<<3072, 256, 0, stream>>>(xbf, pw, bx, xp0);
  gru_steps<<<64,   256, 0, stream>>>(pw, xp0, bx, hf32, hbf, rhb, zb, g2, h1h,
                                      out_hidden, bar);
  gru_out  <<<2048, 256, 0, stream>>>(h1h, pw, bo, out);
}
